// LSTMPredictor_23149873725473
// MI455X (gfx1250) — matrix operand order repaired and checked by reading, not compiled, not run on hardware
//
#include <hip/hip_runtime.h>
#include <cstdint>
#include <cstddef>

// LSTMPredictor: B=256, T=1024, H=64, L=5, future=0
#define B_ 256
#define T_ 1024
#define H_ 64
#define G_ 256   // 4*H (gate dim, PyTorch order i,f,g,o)
#define NL_ 5

typedef __attribute__((ext_vector_type(16))) __bf16 v16bf;
typedef __attribute__((ext_vector_type(8)))  __bf16 v8bf;
typedef __attribute__((ext_vector_type(2)))  __bf16 v2bf;
typedef __attribute__((ext_vector_type(8)))  float  v8f;

union F16x { v16bf v; v8bf h[2]; __bf16 e[16]; };
union F8x  { v8f  v; float e[8]; };
union F2x  { v2bf v; __bf16 e[2]; };

__device__ __forceinline__ v8f wmma_bf16(v16bf a, v16bf b, v8f c) {
  // D = A(16x32 bf16) x B(32x16 bf16) + C(16x16 f32)
  return __builtin_amdgcn_wmma_f32_16x16x32_bf16(
      /*neg_a=*/false, a, /*neg_b=*/false, b,
      /*c_mod=*/(short)0, c, /*reuse_a=*/false, /*reuse_b=*/false);
}

// Fast activations on the serial recurrence path: v_exp_f32 + v_rcp_f32
// (TRANS ops) instead of IEEE divide / libm tanh. rcp(inf)=0 gives correct
// saturation at the extremes.
__device__ __forceinline__ float fast_sig(float x) {
  return __builtin_amdgcn_rcpf(1.0f + __expf(-x));
}
__device__ __forceinline__ float fast_tanh(float x) {
  return 2.0f * __builtin_amdgcn_rcpf(1.0f + __expf(-2.0f * x)) - 1.0f;
}

// ---------------------------------------------------------------------------
// Prep kernels
// ---------------------------------------------------------------------------
__global__ void f32_to_bf16_kernel(const float* __restrict__ src,
                                   __bf16* __restrict__ dst, int n) {
  int i = blockIdx.x * blockDim.x + threadIdx.x;
  if (i < n) dst[i] = (__bf16)src[i];
}

__global__ void bias_combine_kernel(const float* __restrict__ a,
                                    const float* __restrict__ b,
                                    float* __restrict__ dst, int n) {
  int i = blockIdx.x * blockDim.x + threadIdx.x;
  if (i < n) dst[i] = a[i] + b[i];
}

// ---------------------------------------------------------------------------
// One LSTM layer: grid = 16 batch-tiles (batch rows are independent -> no
// grid-wide sync), block = 512 (16 waves). Each wave owns ONE 16-col gate
// tile. Per timestep:
//   gates[16,256] = x_t @ W_ih^T + h_{t-1} @ W_hh^T + (b_ih + b_hh)  via WMMA
//   then c,h elementwise update (2 cells/thread, c in registers).
// hcur and the gate buffer are double-buffered -> only the two TRUE
// dependencies (gates visible, new h visible) need barriers: 2 per step.
// Time loop is unrolled by 2 with NAMED register fragment buffers so the
// x_t software pipeline (loads issued one step ahead) stays in VGPRs.
// Layer 0 (IN_SCALAR): x[b,t]*w_ih0[n] is added in the elementwise phase.
// ---------------------------------------------------------------------------
template <bool IN_SCALAR>
__global__ __launch_bounds__(512) void lstm_layer_kernel(
    const float*  __restrict__ x_scalar,  // [B,T]      (IN_SCALAR)
    const float*  __restrict__ wih0,      // [256]      (IN_SCALAR)
    const __bf16* __restrict__ xin,       // [B,T,H]    (!IN_SCALAR)
    const __bf16* __restrict__ wih,       // [256][64]  (!IN_SCALAR)
    const __bf16* __restrict__ whh,       // [256][64]
    const float*  __restrict__ bias,      // [256]  b_ih + b_hh
    __bf16*       __restrict__ hout)      // [B,T,H]
{
  __shared__ __align__(16) float  gbufT[2][G_][16];  // [buf][gate col][row]
  __shared__ __align__(16) __bf16 hcur[2][16][H_];   // [buf][row][hcol]

  const int tid   = threadIdx.x;
  const int wave  = tid >> 5;          // 0..15 -> N-tile index
  const int lane  = tid & 31;
  const int half  = lane >> 4;
  const int lrow  = lane & 15;
  const int bbase = blockIdx.x * 16;

  const int n = wave * 16 + lrow;      // this lane's gate column

  // ---- Loop-invariant B-matrix fragments (weights) in registers ----
  // B 32x16 bf16 layout: lane holds col N=lrow, K = half*8 + {0..7} then 16 + half*8 + {0..7}.
  F16x whhF0, whhF1, wihF0, wihF1;
  {
    const __bf16* rp = whh + (size_t)n * H_ + half * 8;
    whhF0.h[0] = *(const v8bf*)(rp);       whhF0.h[1] = *(const v8bf*)(rp + 16);
    whhF1.h[0] = *(const v8bf*)(rp + 32);  whhF1.h[1] = *(const v8bf*)(rp + 48);
  }
  if constexpr (!IN_SCALAR) {
    const __bf16* rp = wih + (size_t)n * H_ + half * 8;
    wihF0.h[0] = *(const v8bf*)(rp);       wihF0.h[1] = *(const v8bf*)(rp + 16);
    wihF1.h[0] = *(const v8bf*)(rp + 32);  wihF1.h[1] = *(const v8bf*)(rp + 48);
  }
  const float biasn = bias[n];

  // ---- Elementwise cell mapping: fixed per thread -> c lives in regs ----
  const int erow = tid >> 5;           // 0..15 batch row in tile (== wave)
  const int ecb  = lane * 2;           // 2 consecutive h columns
  float creg[2] = {0.f, 0.f};
  // Layer-0 input-projection coefficients for this thread's cells
  float wc[2][4] = {};
  if constexpr (IN_SCALAR) {
#pragma unroll
    for (int j = 0; j < 2; ++j) {
      const int hc = ecb + j;
      wc[j][0] = wih0[hc];
      wc[j][1] = wih0[64 + hc];
      wc[j][2] = wih0[128 + hc];
      wc[j][3] = wih0[192 + hc];
    }
  }

  { // h_0 = 0 in buffer 0
    F2x z; z.e[0] = (__bf16)0.0f; z.e[1] = (__bf16)0.0f;
    *(v2bf*)(&hcur[0][erow][ecb]) = z.v;
  }
  __syncthreads();

  const size_t arow  = (size_t)(bbase + lrow) * T_;  // A-frag row (time units)
  const size_t xrowb = (size_t)(bbase + erow) * T_;  // layer-0 x row

  // ---- x software pipeline in NAMED registers (no dynamic indexing) ----
  F16x xA0, xA1, xB0, xB1;   // ping (A) / pong (B), two K-steps each
  if constexpr (!IN_SCALAR) {
    const __bf16* rp = xin + arow * H_;   // t = 0 -> A
    xA0.h[0] = *(const v8bf*)(rp + half * 8);
    xA0.h[1] = *(const v8bf*)(rp + 16 + half * 8);
    xA1.h[0] = *(const v8bf*)(rp + 32 + half * 8);
    xA1.h[1] = *(const v8bf*)(rp + 32 + 16 + half * 8);
  }

  // One recurrence step. rb is a literal (0/1) at every call site; xu* are
  // this step's x fragments, xf* get filled for step t+1.
  auto step = [&](int t, int rb, F16x& xu0, F16x& xu1, F16x& xf0, F16x& xf1)
      __attribute__((always_inline)) {
    const int wb = rb ^ 1;

    // Phase A0: issue next-step independent loads first
    if constexpr (!IN_SCALAR) {
      if (t + 1 < T_) {
        const __bf16* rpn = xin + (arow + t + 1) * H_;
        xf0.h[0] = *(const v8bf*)(rpn + half * 8);
        xf0.h[1] = *(const v8bf*)(rpn + 16 + half * 8);
        xf1.h[0] = *(const v8bf*)(rpn + 32 + half * 8);
        xf1.h[1] = *(const v8bf*)(rpn + 32 + 16 + half * 8);
        if (t + 2 < T_) __builtin_prefetch(rpn + H_, 0, 1);
      }
    }
    float xv = 0.0f;
    if constexpr (IN_SCALAR) xv = x_scalar[xrowb + t];  // in flight over WMMA

    // Phase A1: h fragments + WMMA (reads hcur[rb])
    // A-frag (16x32 bf16): lane holds row M=lrow,
    // K = half*8 + {0..7} then 16 + half*8 + {0..7}; kstep adds 32.
    F16x hF0, hF1;
    {
      const __bf16* rp0 = &hcur[rb][lrow][0];
      hF0.h[0] = *(const v8bf*)(rp0 + half * 8);
      hF0.h[1] = *(const v8bf*)(rp0 + 16 + half * 8);
      hF1.h[0] = *(const v8bf*)(rp0 + 32 + half * 8);
      hF1.h[1] = *(const v8bf*)(rp0 + 32 + 16 + half * 8);
    }

    F8x acc;
#pragma unroll
    for (int i = 0; i < 8; ++i) acc.e[i] = biasn;
    acc.v = wmma_bf16(hF0.v, whhF0.v, acc.v);       // recurrent chain, depth 2
    acc.v = wmma_bf16(hF1.v, whhF1.v, acc.v);

    if constexpr (!IN_SCALAR) {
      F8x accx;                                      // independent x chain
#pragma unroll
      for (int i = 0; i < 8; ++i) accx.e[i] = 0.0f;
      accx.v = wmma_bf16(xu0.v, wihF0.v, accx.v);    // operands preloaded
      accx.v = wmma_bf16(xu1.v, wihF1.v, accx.v);    //   last step: no wait
      acc.v = acc.v + accx.v;
    }

    // Phase B: store pre-activation gates into gbufT[rb]
    // C/D layout: elem r -> row (r + 8*half), col n.
    {
      float* p = &gbufT[rb][n][8 * half];
#pragma unroll
      for (int r = 0; r < 8; ++r) p[r] = acc.e[r];
    }

    __syncthreads();  // TRUE dep: gates(t) visible to all waves

    // Phase C: elementwise LSTM cell (2 cells/thread)
    {
      F2x hv;
#pragma unroll
      for (int j = 0; j < 2; ++j) {
        const int hc = ecb + j;
        float gi = gbufT[rb][hc       ][erow];
        float gf = gbufT[rb][ 64 + hc ][erow];
        float gg = gbufT[rb][128 + hc ][erow];
        float go = gbufT[rb][192 + hc ][erow];
        if constexpr (IN_SCALAR) {
          gi = __builtin_fmaf(xv, wc[j][0], gi);
          gf = __builtin_fmaf(xv, wc[j][1], gf);
          gg = __builtin_fmaf(xv, wc[j][2], gg);
          go = __builtin_fmaf(xv, wc[j][3], go);
        }
        const float c = fast_sig(gf) * creg[j] + fast_sig(gi) * fast_tanh(gg);
        creg[j] = c;
        const float h = fast_sig(go) * fast_tanh(c);
        hv.e[j] = (__bf16)h;
      }
      *(v2bf*)(&hcur[wb][erow][ecb]) = hv.v;               // one ds_store_b32
      *(v2bf*)(hout + (xrowb + t) * H_ + ecb) = hv.v;
    }

    __syncthreads();  // TRUE dep: h(t) in hcur[wb] visible for step t+1
  };

  for (int t = 0; t < T_; t += 2) {
    step(t,     0, xA0, xA1, xB0, xB1);   // even step: use A, fill B
    step(t + 1, 1, xB0, xB1, xA0, xA1);   // odd  step: use B, fill A
  }
}

// ---------------------------------------------------------------------------
// Head: out = relu(relu(h) @ W1^T + b1) @ W2^T + b2, fully parallel over B*T.
// One 16-row tile per wave; 8 waves per block; grid = B*T/128 = 2048.
// ---------------------------------------------------------------------------
__global__ __launch_bounds__(256) void head_kernel(
    const __bf16* __restrict__ act,  // [B*T, 64] (h of last LSTM layer)
    const __bf16* __restrict__ w1,   // [64][64]
    const float*  __restrict__ b1,   // [64]
    const float*  __restrict__ w2,   // [64]  (W2 row)
    const float*  __restrict__ b2,   // [1]
    float*        __restrict__ out)  // [B*T]
{
  __shared__ __align__(16) float hbuf[8][16][H_];

  const int tid  = threadIdx.x;
  const int wave = tid >> 5;
  const int lane = tid & 31;
  const int half = lane >> 4;
  const int lrow = lane & 15;
  const size_t rowbase = ((size_t)blockIdx.x * 8 + wave) * 16;

  // W1 B-fragments: 4 N-tiles x 2 K-steps
  F16x w1F[4][2];
#pragma unroll
  for (int nt = 0; nt < 4; ++nt) {
    const int nn = nt * 16 + lrow;
#pragma unroll
    for (int s = 0; s < 2; ++s) {
      const __bf16* rp = w1 + (size_t)nn * H_ + s * 32 + half * 8;
      w1F[nt][s].h[0] = *(const v8bf*)(rp);
      w1F[nt][s].h[1] = *(const v8bf*)(rp + 16);
    }
  }

  // A-fragments of relu(h)
  F16x aF[2];
  const __bf16* rp = act + (rowbase + lrow) * H_;
#pragma unroll
  for (int s = 0; s < 2; ++s) {
    aF[s].h[0] = *(const v8bf*)(rp + s * 32 + half * 8);
    aF[s].h[1] = *(const v8bf*)(rp + s * 32 + 16 + half * 8);
#pragma unroll
    for (int i = 0; i < 16; ++i) {
      const float v = (float)aF[s].e[i];
      aF[s].e[i] = (__bf16)(v > 0.0f ? v : 0.0f);
    }
  }

#pragma unroll
  for (int nt = 0; nt < 4; ++nt) {
    F8x acc;
    const float bn = b1[nt * 16 + lrow];
#pragma unroll
    for (int i = 0; i < 8; ++i) acc.e[i] = bn;
    acc.v = wmma_bf16(aF[0].v, w1F[nt][0].v, acc.v);
    acc.v = wmma_bf16(aF[1].v, w1F[nt][1].v, acc.v);
#pragma unroll
    for (int r = 0; r < 8; ++r) {
      float v = acc.e[r];
      hbuf[wave][r + 8 * half][nt * 16 + lrow] = (v > 0.0f ? v : 0.0f);
    }
  }
  __syncthreads();

  if (lane < 16) {  // final [*,64] x [64,1] dot
    float s = b2[0];
#pragma unroll
    for (int k = 0; k < H_; ++k) s += hbuf[wave][lrow][k] * w2[k];
    out[rowbase + lrow] = s;
  }
}

// ---------------------------------------------------------------------------
// Host orchestration
// ---------------------------------------------------------------------------
extern "C" void kernel_launch(void* const* d_in, const int* in_sizes, int n_in,
                              void* d_out, int out_size, void* d_ws, size_t ws_size,
                              hipStream_t stream) {
  const float* x      = (const float*)d_in[0];  // [B,T,1]
  const float* w_ih0  = (const float*)d_in[1];  // [256,1]
  const float* w_ih_r = (const float*)d_in[2];  // [4,256,64]
  const float* w_hh   = (const float*)d_in[3];  // [5,256,64]
  const float* b_ih   = (const float*)d_in[4];  // [5,256]
  const float* b_hh   = (const float*)d_in[5];  // [5,256]
  const float* W1     = (const float*)d_in[6];  // [64,64]
  const float* b1     = (const float*)d_in[7];  // [64]
  const float* W2     = (const float*)d_in[8];  // [1,64]
  const float* b2     = (const float*)d_in[9];  // [1]
  // d_in[10] = future (0 in eval harness) -> single forward pass
  float* out = (float*)d_out;                   // [B,T,1]

  // Workspace layout (all offsets 256B-aligned)
  char* ws = (char*)d_ws;
  const size_t ACT_BYTES = (size_t)B_ * T_ * H_ * sizeof(__bf16);  // 32 MiB
  __bf16* actA  = (__bf16*)(ws);
  __bf16* actB  = (__bf16*)(ws + ACT_BYTES);
  __bf16* whhB  = (__bf16*)(ws + 2 * ACT_BYTES);                       // 5*256*64
  __bf16* wihB  = (__bf16*)(ws + 2 * ACT_BYTES + 163840);              // 4*256*64
  __bf16* w1B   = (__bf16*)(ws + 2 * ACT_BYTES + 163840 + 131072);     // 64*64
  float*  biasC = (float*) (ws + 2 * ACT_BYTES + 163840 + 131072 + 8192); // 5*256

  // Weight prep (tiny)
  f32_to_bf16_kernel<<<(NL_ * G_ * H_ + 255) / 256, 256, 0, stream>>>(w_hh, whhB, NL_ * G_ * H_);
  f32_to_bf16_kernel<<<((NL_ - 1) * G_ * H_ + 255) / 256, 256, 0, stream>>>(w_ih_r, wihB, (NL_ - 1) * G_ * H_);
  f32_to_bf16_kernel<<<(H_ * H_ + 255) / 256, 256, 0, stream>>>(W1, w1B, H_ * H_);
  bias_combine_kernel<<<(NL_ * G_ + 255) / 256, 256, 0, stream>>>(b_ih, b_hh, biasC, NL_ * G_);

  // Layer 0 (scalar input)
  lstm_layer_kernel<true><<<B_ / 16, 512, 0, stream>>>(
      x, w_ih0, nullptr, nullptr, whhB, biasC, actA);

  // Layers 1..4 (fused input projection + recurrence), ping-pong activations
  __bf16* pin  = actA;
  __bf16* pout = actB;
  for (int l = 1; l < NL_; ++l) {
    lstm_layer_kernel<false><<<B_ / 16, 512, 0, stream>>>(
        nullptr, nullptr,
        pin, wihB + (size_t)(l - 1) * G_ * H_,
        whhB + (size_t)l * G_ * H_,
        biasC + (size_t)l * G_,
        pout);
    __bf16* tmp = pin; pin = pout; pout = tmp;
  }
  // After 4 swaps, `pin` holds the last layer's output (actA).

  // MLP head
  head_kernel<<<(B_ * T_) / 128, 256, 0, stream>>>(pin, w1B, b1, W2, b2, out);
}
